// SwinTransformerBlock_7516192768245
// MI455X (gfx1250) — compile-verified, matrix operand order checked by reading, not run on hardware
//
#include <hip/hip_runtime.h>
#include <hip/hip_bf16.h>

// ---------------------------------------------------------------------------
// Swin-3D block for MI455X (gfx1250): f16 WMMA GEMMs + fp32 spine.
// Dims hardcoded from reference: B=2,T=8,H=56,W=56,C=384,NH=12,hd=32,
// window 4x7x7 -> N=196, B_=256 windows, hid=1536, L=25088.
// ---------------------------------------------------------------------------

typedef __attribute__((ext_vector_type(16))) _Float16 v16h;
typedef __attribute__((ext_vector_type(8)))  _Float16 v8h;
typedef __attribute__((ext_vector_type(8)))  float    v8f;

#define CDIM   384
#define ROWS   50176            // B * L  ==  B_ * N
#define LLEN   25088
#define NWIN   196
#define NHEADS 12
#define HID    1536
#define QK_SCALE 0.17677669529663687f   // 32^-0.5
#define VPAD   224              // padded key length (7 * 32)

// ---------------- WMMA wrapper ---------------------------------------------
static __device__ inline v8f wmma16(v16h a, v16h b, v8f c) {
#if defined(__gfx1250__)
  return __builtin_amdgcn_wmma_f32_16x16x32_f16(false, a, false, b, (short)0, c,
                                                false, false);
#else
  (void)a; (void)b;
  return c;
#endif
}

// A/B fragment loader, row-major f16 source, 16x32 tile at (row0, k0).
// Layout (ISA 7.12.2, 16-bit A 16x32): lanes 0-15 hold K 0..7 & 16..23,
// lanes 16-31 hold K 8..15 & 24..31; row = row0 + (lane&15).
static __device__ inline v16h ldfrag(const _Float16* base, long ld, long row0,
                                     int k0, int lane) {
  const _Float16* p = base + (row0 + (lane & 15)) * ld + k0 + ((lane >> 4) << 3);
  v8h lo = *(const v8h*)p;
  v8h hi = *(const v8h*)(p + 16);
  return __builtin_shufflevector(lo, hi, 0,1,2,3,4,5,6,7,8,9,10,11,12,13,14,15);
}

static __device__ inline v16h ldfrag_clamp(const _Float16* base, long ld,
                                           int row0, int rmax, int k0, int lane) {
  int r = row0 + (lane & 15);
  if (r > rmax) r = rmax;
  const _Float16* p = base + (long)r * ld + k0 + ((lane >> 4) << 3);
  v8h lo = *(const v8h*)p;
  v8h hi = *(const v8h*)(p + 16);
  return __builtin_shufflevector(lo, hi, 0,1,2,3,4,5,6,7,8,9,10,11,12,13,14,15);
}

// Per-wave 64x32 GEMM core: A row-major [M][K], W row-major [N][K] (so B
// columns are W rows -> contiguous).  acc[4][2] of 16x16 f32 tiles.
template <int K>
static __device__ inline void gemm_core(const _Float16* __restrict__ A,
                                        const _Float16* __restrict__ W,
                                        long m0, long n0, int lane,
                                        v8f (&acc)[4][2]) {
  for (int k0 = 0; k0 < K; k0 += 32) {
    v16h af[4], bf[2];
#pragma unroll
    for (int i = 0; i < 4; ++i) af[i] = ldfrag(A, K, m0 + i * 16, k0, lane);
#pragma unroll
    for (int j = 0; j < 2; ++j) bf[j] = ldfrag(W, K, n0 + j * 16, k0, lane);
#pragma unroll
    for (int i = 0; i < 4; ++i)
#pragma unroll
      for (int j = 0; j < 2; ++j) acc[i][j] = wmma16(af[i], bf[j], acc[i][j]);
  }
}

// ---------------- small helpers --------------------------------------------
__global__ __launch_bounds__(256) void cvt_f16_kernel(const float* __restrict__ s,
                                                      _Float16* __restrict__ d,
                                                      int n) {
  int i = blockIdx.x * 256 + threadIdx.x;
  if (i < n) d[i] = (_Float16)s[i];
}

// ---------------- LN1 + quirky reshape/roll/window scatter -----------------
// y = LN(x) in (B,L,C); memory reinterpreted per batch as (C,H,W,T), permuted
// to (T,H,W,C), rolled by (-2,-3,-3), window-partitioned into (B_,N,C) f16.
__global__ __launch_bounds__(256) void ln1_scatter_kernel(
    const float* __restrict__ x, const float* __restrict__ g,
    const float* __restrict__ b, _Float16* __restrict__ hwin) {
  int wave = threadIdx.x >> 5, lane = threadIdx.x & 31;
  long r = (long)blockIdx.x * 8 + wave;          // row in (B*L)
  const float* xr = x + r * CDIM;
  float vals[12];
  float s = 0.f;
#pragma unroll
  for (int i = 0; i < 12; ++i) { vals[i] = xr[lane + 32 * i]; s += vals[i]; }
#pragma unroll
  for (int o = 16; o >= 1; o >>= 1) s += __shfl_xor(s, o, 32);
  float mean = s * (1.0f / CDIM);
  float vs = 0.f;
#pragma unroll
  for (int i = 0; i < 12; ++i) { float d0 = vals[i] - mean; vs += d0 * d0; }
#pragma unroll
  for (int o = 16; o >= 1; o >>= 1) vs += __shfl_xor(vs, o, 32);
  float rs = rsqrtf(vs * (1.0f / CDIM) + 1e-5f);

  int bi = (int)(r / LLEN);
  int l  = (int)(r % LLEN);
#pragma unroll
  for (int i = 0; i < 12; ++i) {
    int c = lane + 32 * i;
    float y = (vals[i] - mean) * rs * g[c] + b[c];
    // flat position within batch interpreted as (C,H,W,T)
    long p  = (long)l * CDIM + c;
    int cimg = (int)(p / LLEN);
    int rem  = (int)(p % LLEN);
    int hh = rem / 448;
    int r2 = rem % 448;
    int ww = r2 >> 3;
    int t  = r2 & 7;
    // roll by (-2,-3,-3)
    int t2 = (t + 6) & 7;
    int h2 = hh + 53; if (h2 >= 56) h2 -= 56;
    int w2 = ww + 53; if (w2 >= 56) w2 -= 56;
    int wi = ((bi * 2 + (t2 >> 2)) * 8 + h2 / 7) * 8 + w2 / 7;
    int n  = (t2 & 3) * 49 + (h2 % 7) * 7 + (w2 % 7);
    hwin[((long)wi * NWIN + n) * CDIM + cimg] = (_Float16)y;
  }
}

// ---------------- LN2 (plain, x2 f32 -> h2 f16) ----------------------------
__global__ __launch_bounds__(256) void ln2_kernel(
    const float* __restrict__ x2, const float* __restrict__ g,
    const float* __restrict__ b, _Float16* __restrict__ h2) {
  int wave = threadIdx.x >> 5, lane = threadIdx.x & 31;
  long r = (long)blockIdx.x * 8 + wave;
  const float* xr = x2 + r * CDIM;
  float vals[12];
  float s = 0.f;
#pragma unroll
  for (int i = 0; i < 12; ++i) { vals[i] = xr[lane + 32 * i]; s += vals[i]; }
#pragma unroll
  for (int o = 16; o >= 1; o >>= 1) s += __shfl_xor(s, o, 32);
  float mean = s * (1.0f / CDIM);
  float vs = 0.f;
#pragma unroll
  for (int i = 0; i < 12; ++i) { float d0 = vals[i] - mean; vs += d0 * d0; }
#pragma unroll
  for (int o = 16; o >= 1; o >>= 1) vs += __shfl_xor(vs, o, 32);
  float rs = rsqrtf(vs * (1.0f / CDIM) + 1e-5f);
#pragma unroll
  for (int i = 0; i < 12; ++i) {
    int c = lane + 32 * i;
    h2[r * CDIM + c] = (_Float16)((vals[i] - mean) * rs * g[c] + b[c]);
  }
}

// ---------------- QKV GEMM (M=50176, N=1152, K=384) ------------------------
__global__ __launch_bounds__(256) void gemm_qkv_kernel(
    const _Float16* __restrict__ A, const _Float16* __restrict__ W,
    const float* __restrict__ bias, _Float16* __restrict__ qb,
    _Float16* __restrict__ kb, _Float16* __restrict__ vt) {
  int lane = threadIdx.x & 31, wave = threadIdx.x >> 5;
  long m0 = (long)blockIdx.x * 128 + (wave & 1) * 64;
  long n0 = (long)blockIdx.y * 128 + (wave >> 1) * 32;
  v8f acc[4][2] = {};
  gemm_core<CDIM>(A, W, m0, n0, lane, acc);
#pragma unroll
  for (int mt = 0; mt < 4; ++mt)
#pragma unroll
    for (int nt = 0; nt < 2; ++nt)
#pragma unroll
      for (int v = 0; v < 8; ++v) {
        long m = m0 + mt * 16 + v + ((lane >> 4) << 3);
        int  j = (int)n0 + nt * 16 + (lane & 15);
        float val = acc[mt][nt][v] + bias[j];
        int which = j / CDIM;
        int head  = (j % CDIM) >> 5;
        int d     = j & 31;
        long wi = m / NWIN;
        int  n  = (int)(m % NWIN);
        long wh = wi * NHEADS + head;
        if (which == 0) {
          qb[(wh * NWIN + n) * 32 + d] = (_Float16)(val * QK_SCALE);
        } else if (which == 1) {
          kb[(wh * NWIN + n) * 32 + d] = (_Float16)val;
        } else {
          vt[(wh * 32 + d) * VPAD + n] = (_Float16)val;   // transposed, padded
        }
      }
}

// ---------------- Attention: one block per (window, head) ------------------
__global__ __launch_bounds__(256) void attn_kernel(
    const _Float16* __restrict__ qb, const _Float16* __restrict__ kb,
    const _Float16* __restrict__ vt, const float* __restrict__ rpb,
    const int* __restrict__ relidx, const float* __restrict__ mask,
    _Float16* __restrict__ attout) {
  __shared__ float sS[32][225];
  int bx = blockIdx.x;
  int wi = bx / NHEADS, head = bx % NHEADS;
  int lane = threadIdx.x & 31, wave = threadIdx.x >> 5;
  long wh = (long)wi * NHEADS + head;
  const _Float16* q = qb + wh * NWIN * 32;
  const _Float16* k = kb + wh * NWIN * 32;
  const _Float16* vv = vt + wh * 32 * VPAD;
  const float* mk = mask + (long)(wi & 127) * NWIN * NWIN;

  for (int rb = 0; rb < 7; ++rb) {
    int m0 = rb * 32;
    // ---- S = q k^T (+bias +mask) for rows m0..m0+31, cols 0..223 ----
    for (int tile = wave; tile < 28; tile += 8) {
      int mt = tile / 14, nt = tile % 14;
      v16h af = ldfrag_clamp(q, 32, m0 + mt * 16, NWIN - 1, 0, lane);
      v16h bf = ldfrag_clamp(k, 32, nt * 16,      NWIN - 1, 0, lane);
      v8f c = {};
      c = wmma16(af, bf, c);
#pragma unroll
      for (int v = 0; v < 8; ++v) {
        int lr = mt * 16 + v + ((lane >> 4) << 3);   // local row 0..31
        int n  = nt * 16 + (lane & 15);              // key col 0..223
        float s;
        if (n >= NWIN) {
          s = -1e30f;
        } else {
          int mg = m0 + lr; if (mg > NWIN - 1) mg = NWIN - 1;
          s = c[v] + rpb[relidx[mg * NWIN + n] * NHEADS + head]
                   + mk[(long)mg * NWIN + n];
        }
        sS[lr][n] = s;
      }
    }
    __syncthreads();
    // ---- softmax over rows (4 rows per wave) ----
    for (int rr = 0; rr < 4; ++rr) {
      int lr = wave * 4 + rr;
      if (m0 + lr < NWIN) {
        float mx = -1e30f;
        for (int c = lane; c < VPAD; c += 32) mx = fmaxf(mx, sS[lr][c]);
#pragma unroll
        for (int o = 16; o >= 1; o >>= 1) mx = fmaxf(mx, __shfl_xor(mx, o, 32));
        float sum = 0.f;
        for (int c = lane; c < VPAD; c += 32) {
          float e = __expf(sS[lr][c] - mx);
          sS[lr][c] = e;
          sum += e;
        }
#pragma unroll
        for (int o = 16; o >= 1; o >>= 1) sum += __shfl_xor(sum, o, 32);
        float inv = 1.0f / sum;
        for (int c = lane; c < VPAD; c += 32) sS[lr][c] *= inv;
      }
    }
    __syncthreads();
    // ---- O = P V  (2x2 tiles of 16x16 over 4 waves, K = 224) ----
    if (wave < 4) {
      int mt = wave >> 1, nt = wave & 1;
      v8f o = {};
      for (int k0 = 0; k0 < VPAD; k0 += 32) {
        // P fragment from LDS (f32 -> f16 on the fly)
        int pr = mt * 16 + (lane & 15);
        int kbs = k0 + ((lane >> 4) << 3);
        const float* pp = &sS[pr][kbs];
        v16h pf;
#pragma unroll
        for (int j = 0; j < 8; ++j) {
          pf[j]     = (_Float16)pp[j];
          pf[j + 8] = (_Float16)pp[j + 16];
        }
        v16h vf = ldfrag(vv, VPAD, nt * 16, k0, lane);
        o = wmma16(pf, vf, o);
      }
#pragma unroll
      for (int v = 0; v < 8; ++v) {
        int m = m0 + mt * 16 + v + ((lane >> 4) << 3);
        int d = nt * 16 + (lane & 15);
        if (m < NWIN)
          attout[((long)wi * NWIN + m) * CDIM + head * 32 + d] = (_Float16)o[v];
      }
    }
    __syncthreads();
  }
}

// ---------------- Proj GEMM + window-reverse/roll + residual ---------------
__global__ __launch_bounds__(256) void gemm_proj_kernel(
    const _Float16* __restrict__ A, const _Float16* __restrict__ W,
    const float* __restrict__ bias, const float* __restrict__ x,
    float* __restrict__ x2) {
  int lane = threadIdx.x & 31, wave = threadIdx.x >> 5;
  long m0 = (long)blockIdx.x * 128 + (wave & 1) * 64;
  long n0 = (long)blockIdx.y * 128 + (wave >> 1) * 32;
  v8f acc[4][2] = {};
  gemm_core<CDIM>(A, W, m0, n0, lane, acc);
#pragma unroll
  for (int mt = 0; mt < 4; ++mt)
#pragma unroll
    for (int nt = 0; nt < 2; ++nt)
#pragma unroll
      for (int v = 0; v < 8; ++v) {
        long m = m0 + mt * 16 + v + ((lane >> 4) << 3);
        int  c = (int)n0 + nt * 16 + (lane & 15);
        float val = acc[mt][nt][v] + bias[c];
        int wi = (int)(m / NWIN), n = (int)(m % NWIN);
        int bi = wi >> 7, rem = wi & 127;
        int tb = rem >> 6, hb = (rem >> 3) & 7, wb = rem & 7;
        int dt = n / 49, r2 = n % 49, dh = r2 / 7, dw = r2 % 7;
        // quirky window reverse: (b,tb,hb,dt,wb,dh,dw) flatten -> (T,H,W)
        int f = ((((tb * 8 + hb) * 4 + dt) * 8 + wb) * 7 + dh) * 7 + dw;
        int t  = f / 3136;
        int f2 = f % 3136;
        int ho = f2 / 56, wo = f2 % 56;
        t = (t + 2) & 7;
        ho += 3; if (ho >= 56) ho -= 56;
        wo += 3; if (wo >= 56) wo -= 56;
        long l = (long)t * 3136 + ho * 56 + wo;
        long idx = ((long)bi * LLEN + l) * CDIM + c;
        x2[idx] = x[idx] + val;
      }
}

// ---------------- FC1 GEMM + exact GELU ------------------------------------
__global__ __launch_bounds__(256) void gemm_fc1_kernel(
    const _Float16* __restrict__ A, const _Float16* __restrict__ W,
    const float* __restrict__ bias, _Float16* __restrict__ out) {
  int lane = threadIdx.x & 31, wave = threadIdx.x >> 5;
  long m0 = (long)blockIdx.x * 128 + (wave & 1) * 64;
  long n0 = (long)blockIdx.y * 128 + (wave >> 1) * 32;
  v8f acc[4][2] = {};
  gemm_core<CDIM>(A, W, m0, n0, lane, acc);
#pragma unroll
  for (int mt = 0; mt < 4; ++mt)
#pragma unroll
    for (int nt = 0; nt < 2; ++nt)
#pragma unroll
      for (int v = 0; v < 8; ++v) {
        long m = m0 + mt * 16 + v + ((lane >> 4) << 3);
        int  c = (int)n0 + nt * 16 + (lane & 15);
        float z = acc[mt][nt][v] + bias[c];
        float ge = 0.5f * z * (1.0f + erff(z * 0.7071067811865476f));
        out[m * HID + c] = (_Float16)ge;
      }
}

// ---------------- FC2 GEMM + residual -> d_out (fp32) ----------------------
__global__ __launch_bounds__(256) void gemm_fc2_kernel(
    const _Float16* __restrict__ A, const _Float16* __restrict__ W,
    const float* __restrict__ bias, const float* __restrict__ x2,
    float* __restrict__ out) {
  int lane = threadIdx.x & 31, wave = threadIdx.x >> 5;
  long m0 = (long)blockIdx.x * 128 + (wave & 1) * 64;
  long n0 = (long)blockIdx.y * 128 + (wave >> 1) * 32;
  v8f acc[4][2] = {};
  gemm_core<HID>(A, W, m0, n0, lane, acc);
#pragma unroll
  for (int mt = 0; mt < 4; ++mt)
#pragma unroll
    for (int nt = 0; nt < 2; ++nt)
#pragma unroll
      for (int v = 0; v < 8; ++v) {
        long m = m0 + mt * 16 + v + ((lane >> 4) << 3);
        int  c = (int)n0 + nt * 16 + (lane & 15);
        long idx = m * CDIM + c;
        out[idx] = x2[idx] + acc[mt][nt][v] + bias[c];
      }
}

// ---------------------------------------------------------------------------
extern "C" void kernel_launch(void* const* d_in, const int* in_sizes, int n_in,
                              void* d_out, int out_size, void* d_ws, size_t ws_size,
                              hipStream_t stream) {
  (void)in_sizes; (void)n_in; (void)out_size; (void)ws_size;
  const float* x        = (const float*)d_in[0];
  const float* amask    = (const float*)d_in[1];
  const float* n1g      = (const float*)d_in[2];
  const float* n1b      = (const float*)d_in[3];
  const float* qkv_w    = (const float*)d_in[4];
  const float* qkv_b    = (const float*)d_in[5];
  const float* rpb      = (const float*)d_in[6];
  const float* proj_w   = (const float*)d_in[7];
  const float* proj_b   = (const float*)d_in[8];
  const float* n2g      = (const float*)d_in[9];
  const float* n2b      = (const float*)d_in[10];
  const float* fc1_w    = (const float*)d_in[11];
  const float* fc1_b    = (const float*)d_in[12];
  const float* fc2_w    = (const float*)d_in[13];
  const float* fc2_b    = (const float*)d_in[14];
  const int*   relidx   = (const int*)d_in[15];
  float*       out      = (float*)d_out;

  // ---- workspace layout (bytes) ----
  char* ws = (char*)d_ws;
  _Float16* w16    = (_Float16*)ws;                  // converted weights
  _Float16* qkvW16 = w16;                            // 1152*384
  _Float16* projW16= w16 + 442368;                   // 384*384
  _Float16* fc1W16 = w16 + 589824;                   // 1536*384
  _Float16* fc2W16 = w16 + 1179648;                  // 384*1536
  _Float16* bufA   = (_Float16*)(ws + 3538944);      // 50176*384 halves
  _Float16* bufB   = (_Float16*)(ws + 42074112);
  _Float16* qbuf   = bufB;                           // 50176*384
  _Float16* kbuf   = bufB + 19267584;                // 50176*384
  _Float16* vT     = bufB + 2L * 19267584;           // 3072*32*224 (padded)
  _Float16* fc1out = bufB;                           // reuse: 50176*1536
  float*    x2     = (float*)(ws + 196214784);       // 50176*384 f32
  _Float16* hwin   = bufA;                           // reused: hwin -> attout -> h2
  _Float16* attout = bufA;
  _Float16* h2     = bufA;

  // 0) weight conversion to f16
  cvt_f16_kernel<<<(442368 + 255) / 256, 256, 0, stream>>>(qkv_w, qkvW16, 442368);
  cvt_f16_kernel<<<(147456 + 255) / 256, 256, 0, stream>>>(proj_w, projW16, 147456);
  cvt_f16_kernel<<<(589824 + 255) / 256, 256, 0, stream>>>(fc1_w, fc1W16, 589824);
  cvt_f16_kernel<<<(589824 + 255) / 256, 256, 0, stream>>>(fc2_w, fc2W16, 589824);

  // 1) LN1 + reshape/roll/window scatter  (50176 rows, 1 wave/row)
  ln1_scatter_kernel<<<ROWS / 8, 256, 0, stream>>>(x, n1g, n1b, hwin);

  // 2) QKV GEMM
  gemm_qkv_kernel<<<dim3(ROWS / 128, 1152 / 128), 256, 0, stream>>>(
      hwin, qkvW16, qkv_b, qbuf, kbuf, vT);

  // 3) attention: one block per (window, head)
  attn_kernel<<<256 * NHEADS, 256, 0, stream>>>(qbuf, kbuf, vT, rpb, relidx,
                                                amask, attout);

  // 4) proj GEMM + window reverse + roll + residual -> x2
  gemm_proj_kernel<<<dim3(ROWS / 128, CDIM / 128), 256, 0, stream>>>(
      attout, projW16, proj_b, x, x2);

  // 5) LN2 -> f16
  ln2_kernel<<<ROWS / 8, 256, 0, stream>>>(x2, n2g, n2b, h2);

  // 6) FC1 + GELU
  gemm_fc1_kernel<<<dim3(ROWS / 128, HID / 128), 256, 0, stream>>>(
      h2, fc1W16, fc1_b, fc1out);

  // 7) FC2 + residual -> out
  gemm_fc2_kernel<<<dim3(ROWS / 128, CDIM / 128), 256, 0, stream>>>(
      fc1out, fc2W16, fc2_b, x2, out);
}
